// Experts_33535104647681
// MI455X (gfx1250) — matrix operand order repaired and checked by reading, not compiled, not run on hardware
//
#include <hip/hip_runtime.h>
#include <hip/hip_bf16.h>
#include <math.h>

// MoE grouped FFN: y[e] = GELU(x[e] @ w1[e] + b1[e]) @ w2[e] + b2[e]
// E=8, C=2048, D=2048, F=8192. Compute-bound at bf16 (AI ~820 FLOP/B vs
// ~215 FLOP/B machine balance) -> v_wmma_f32_16x16x32_bf16 with f32 accum.
// h kept in d_ws as bf16 (E*C*F*2 = 256 MiB). GEMM2's A operand is bf16 ->
// staged with GLOBAL_LOAD_ASYNC_TO_LDS_B64 (ASYNCcnt) into a double-buffered
// LDS tile; fp32 operands are software-pipelined through registers with
// on-the-fly cvt_pk_bf16_f32.

#define NE 8
#define NC 2048
#define ND 2048
#define NF 8192

#define BM 128
#define BN 128
#define BK 32
#define BKP 48   // padded LDS k-stride (elements); keeps fragments 16B/32B aligned

typedef __attribute__((ext_vector_type(16))) __bf16 v16bf;
typedef __attribute__((ext_vector_type(8)))  float  v8f;

// Async global->LDS byte copy (8B per lane). Generic LDS pointer's low 32 bits
// are the LDS byte offset (ISA 10.2).
__device__ __forceinline__ void async_copy_b64(const void* g, void* l) {
#if defined(__AMDGCN__)
    unsigned int laddr = (unsigned int)(unsigned long long)l;
    asm volatile("global_load_async_to_lds_b64 %0, %1, off"
                 :: "v"(laddr), "v"(g) : "memory");
#else
    __builtin_memcpy(l, g, 8);
#endif
}
__device__ __forceinline__ void wait_asynccnt0() {
#if defined(__AMDGCN__)
    asm volatile("s_wait_asynccnt 0" ::: "memory");
#endif
}

template<bool A_BF16, bool GELU_BF16OUT>
__global__ __launch_bounds__(256)
void moe_wmma_gemm(const void* __restrict__ Av, const float* __restrict__ B,
                   const float* __restrict__ bias, void* __restrict__ Ov,
                   int M, int N, int K)
{
    __shared__ __bf16 As[2][BM * BKP] __attribute__((aligned(16))); // [m][k], double-buffered
    __shared__ __bf16 Bs[BN * BKP]    __attribute__((aligned(16))); // [n][k] (transposed)

    const int e       = blockIdx.z;
    const int rowBase = blockIdx.y * BM;
    const int colBase = blockIdx.x * BN;

    const size_t eAoff = (size_t)e * M * K;
    const size_t eBoff = (size_t)e * K * N;
    const size_t eOoff = (size_t)e * M * N;

    const float*  Af = (const float*)Av  + eAoff;   // !A_BF16 path (x)
    const __bf16* Ah = (const __bf16*)Av + eAoff;   //  A_BF16 path (h)
    const float*  Be = B    + eBoff;
    const float*  be = bias + (size_t)e * N;

    const int tid   = threadIdx.x;
    const int lane  = tid & 31;
    const int wv    = tid >> 5;
    const int waveM = (wv >> 2) * 64;   // 2 waves along M
    const int waveN = (wv & 3)  * 32;   // 4 waves along N
    const int ln    = lane & 15;
    const int h2    = lane >> 4;

    v8f acc[4][2] = {};                 // 64x32 wave tile = 4x2 16x16 fragments

    float4 regA[4];                     // next A chunk (fp32 path)
    float4 regB[4];                     // next B chunk

    // ---- staging helpers (each thread moves 4 chunks of 4 elements) ----
    auto issueA = [&](int kk, int buf) {
        #pragma unroll
        for (int it = 0; it < 4; ++it) {
            const int c  = tid + 256 * it;
            const int r  = c >> 3;          // 0..127
            const int kc = (c & 7) * 4;     // 0..28
            if (A_BF16) {
                async_copy_b64(Ah + (size_t)(rowBase + r) * K + kk + kc,
                               &As[buf][r * BKP + kc]);
            } else {
                regA[it] = *(const float4*)(Af + (size_t)(rowBase + r) * K + kk + kc);
            }
        }
    };
    auto issueB = [&](int kk) {
        #pragma unroll
        for (int it = 0; it < 4; ++it) {
            const int c  = tid + 256 * it;
            const int k  = c >> 5;          // 0..31
            const int nc = (c & 31) * 4;    // 0..124
            regB[it] = *(const float4*)(Be + (size_t)(kk + k) * N + colBase + nc);
        }
    };
    auto storeA = [&](int buf) {            // fp32 path: convert + contiguous b64 store
        #pragma unroll
        for (int it = 0; it < 4; ++it) {
            const int c  = tid + 256 * it;
            const int r  = c >> 3;
            const int kc = (c & 7) * 4;
            __bf16 t[4] __attribute__((aligned(8)));
            t[0] = (__bf16)regA[it].x; t[1] = (__bf16)regA[it].y;
            t[2] = (__bf16)regA[it].z; t[3] = (__bf16)regA[it].w;
            __builtin_memcpy(&As[buf][r * BKP + kc], t, 8);
        }
    };
    auto storeB = [&]() {                   // convert + transpose into Bs[n][k]
        #pragma unroll
        for (int it = 0; it < 4; ++it) {
            const int c  = tid + 256 * it;
            const int k  = c >> 5;
            const int nc = (c & 31) * 4;
            Bs[(nc + 0) * BKP + k] = (__bf16)regB[it].x;
            Bs[(nc + 1) * BKP + k] = (__bf16)regB[it].y;
            Bs[(nc + 2) * BKP + k] = (__bf16)regB[it].z;
            Bs[(nc + 3) * BKP + k] = (__bf16)regB[it].w;
        }
    };

    // ---- software pipeline: prologue loads chunk 0 ----
    int p = 0;
    issueA(0, 0);
    issueB(0);

    for (int kk = 0; kk < K; kk += BK) {
        // publish chunk kk into LDS
        storeB();
        if (!A_BF16) storeA(p);
        else         wait_asynccnt0();      // async writes into As[p] complete
        __syncthreads();

        // issue chunk kk+BK while computing (async A -> As[p^1]; B -> regs)
        if (kk + BK < K) {
            issueA(kk + BK, p ^ 1);
            issueB(kk + BK);
        }

        // ---- fragments per ISA VGPR layouts ----
        // A 16x32 bf16: element i <-> K = 16*(i>=8) + 8*h2 + (i&7)  (two 16B chunks)
        v16bf afr[4];
        #pragma unroll
        for (int mt = 0; mt < 4; ++mt) {
            const __bf16* q = &As[p][(waveM + mt * 16 + ln) * BKP + 8 * h2];
            __builtin_memcpy(&afr[mt], q, 16);
            __builtin_memcpy((char*)&afr[mt] + 16, q + 16, 16);
        }
        // B 32x16 bf16: element i <-> K = 16*(i>=8) + 8*h2 + (i&7)  (two 16B chunks)
        v16bf bfr[2];
        #pragma unroll
        for (int nt = 0; nt < 2; ++nt) {
            const __bf16* q = &Bs[(waveN + nt * 16 + ln) * BKP + 8 * h2];
            __builtin_memcpy(&bfr[nt], q, 16);
            __builtin_memcpy((char*)&bfr[nt] + 16, q + 16, 16);
        }

        #pragma unroll
        for (int mt = 0; mt < 4; ++mt)
            #pragma unroll
            for (int nt = 0; nt < 2; ++nt)
                acc[mt][nt] = __builtin_amdgcn_wmma_f32_16x16x32_bf16(
                    false, afr[mt], false, bfr[nt],
                    (short)0, acc[mt][nt], false, false);

        __syncthreads();
        p ^= 1;
    }

    // ---- epilogue: C/D layout is VGPR r <-> M = r + 8*h2, N = ln ----
    #pragma unroll
    for (int nt = 0; nt < 2; ++nt) {
        const int col = colBase + waveN + nt * 16 + ln;
        const float bv = be[col];
        #pragma unroll
        for (int mt = 0; mt < 4; ++mt) {
            #pragma unroll
            for (int r = 0; r < 8; ++r) {
                const int row = rowBase + waveM + mt * 16 + r + 8 * h2;
                float v = acc[mt][nt][r] + bv;
                const size_t idx = eOoff + (size_t)row * N + col;
                if (GELU_BF16OUT) {
                    v = 0.5f * v * (1.0f + erff(v * 0.70710678118654752f)); // exact GELU
                    ((__bf16*)Ov)[idx] = (__bf16)v;
                } else {
                    ((float*)Ov)[idx] = v;
                }
            }
        }
    }
}

extern "C" void kernel_launch(void* const* d_in, const int* in_sizes, int n_in,
                              void* d_out, int out_size, void* d_ws, size_t ws_size,
                              hipStream_t stream) {
    (void)in_sizes; (void)n_in; (void)out_size; (void)ws_size;
    const float* x  = (const float*)d_in[0];   // [1, E*C, D]
    const float* w1 = (const float*)d_in[1];   // [E, D, F]
    const float* b1 = (const float*)d_in[2];   // [E, F]
    const float* w2 = (const float*)d_in[3];   // [E, F, D]
    const float* b2 = (const float*)d_in[4];   // [E, D]
    void* h = d_ws;                            // bf16 [E, C, F] = 256 MiB

    dim3 blk(256, 1, 1);

    // GEMM1: h = GELU(x @ w1 + b1)   M=C, N=F, K=D
    dim3 g1(NF / BN, NC / BM, NE);
    moe_wmma_gemm<false, true><<<g1, blk, 0, stream>>>(x, w1, b1, h, NC, NF, ND);

    // GEMM2: y = h @ w2 + b2         M=C, N=D, K=F
    dim3 g2(ND / BN, NC / BM, NE);
    moe_wmma_gemm<true, false><<<g2, blk, 0, stream>>>(h, w2, b2, d_out, NC, ND, NF);
}
